// FixedProductionSplatFlowAttention_38431367364864
// MI455X (gfx1250) — matrix operand order repaired and checked by reading, not compiled, not run on hardware
//
#include <hip/hip_runtime.h>
#include <hip/hip_bf16.h>

typedef __attribute__((ext_vector_type(16))) __bf16 v16bf;
typedef __attribute__((ext_vector_type(8)))  __bf16 v8bf;
typedef __attribute__((ext_vector_type(8)))  float  v8f;
typedef __attribute__((ext_vector_type(4))) unsigned int u32x4;
typedef __attribute__((ext_vector_type(8))) int i32x8;
typedef __attribute__((ext_vector_type(4))) int i32x4;

// ---------------------------------------------------------------------------
// WMMA helper (CDNA5 gfx1250, wave32)
// ---------------------------------------------------------------------------
__device__ __forceinline__ v8f wmma_bf16(v16bf a, v16bf b, v8f c) {
  return __builtin_amdgcn_wmma_f32_16x16x32_bf16(
      /*neg_a=*/false, a, /*neg_b=*/false, b,
      /*c_mod=*/(short)0, c, /*reuse_a=*/false, /*reuse_b=*/false);
}

// A fragment from row-major bf16 [.. x lda]: lane L -> m = m0+(L&15), half=L>>4,
// elems 0..7 -> K=k0+half*8+e, elems 8..15 -> K=k0+16+half*8+(e-8).
__device__ __forceinline__ v16bf load_frag_a(const __bf16* __restrict__ A,
                                             int lda, int m0, int k0, int lane) {
  int m  = m0 + (lane & 15);
  int hf = lane >> 4;
  const __bf16* p = A + (size_t)m * lda + k0 + hf * 8;
  v8bf lo = *(const v8bf*)p;
  v8bf hi = *(const v8bf*)(p + 16);
  return __builtin_shufflevector(lo, hi, 0,1,2,3,4,5,6,7,8,9,10,11,12,13,14,15);
}

// B fragment from transposed-B (Bt row-major [N x K]): lane L -> n = n0+(L&15),
// elems 0..7 -> K=k0+half*8+e, elems 8..15 -> K=k0+16+half*8+(e-8).
__device__ __forceinline__ v16bf load_frag_b(const __bf16* __restrict__ Bt,
                                             int ldb, int n0, int k0, int lane) {
  int n  = n0 + (lane & 15);
  int hf = lane >> 4;
  const __bf16* p = Bt + (size_t)n * ldb + k0 + hf * 8;
  v8bf lo = *(const v8bf*)p;
  v8bf hi = *(const v8bf*)(p + 16);
  return __builtin_shufflevector(lo, hi, 0,1,2,3,4,5,6,7,8,9,10,11,12,13,14,15);
}

// ---------------------------------------------------------------------------
// Tensor Data Mover: async 2-D bf16 tile load (tile_k x tile_rows) into LDS.
// D# built per CDNA5 ISA ch.8: group0 {count=1, lds_addr, global_addr, type=2},
// group1 {data_size=1(2B), tensor dims, tile dims, dim0 stride}, groups2/3 = 0.
// This toolchain declares the 6-arg builtin:
//   (uint32x4 g0, int32x8 g1, int32x4, int32x4, int32x8, i32 cpol)
// ---------------------------------------------------------------------------
__device__ __forceinline__ void tdm_load_tile_bf16(unsigned lds_byte_off,
                                                   const void* gaddr,
                                                   unsigned tile_k, unsigned tile_rows,
                                                   unsigned long long row_stride_elems) {
  unsigned long long ga = (unsigned long long)gaddr;
  u32x4 g0;
  g0[0] = 1u;                                                    // count=1, user D#
  g0[1] = lds_byte_off;                                          // lds_addr
  g0[2] = (unsigned)ga;                                          // global_addr[31:0]
  g0[3] = (unsigned)((ga >> 32) & 0x01FFFFFFull) | (2u << 30);   // addr[56:32] | type=2
  i32x8 g1;
  g1[0] = (int)(1u << 16);                                       // wg_mask=0, data_size=2B
  g1[1] = (int)((tile_k & 0xFFFFu) << 16);                       // tensor_dim0[15:0]
  g1[2] = (int)((tile_k >> 16) | ((tile_rows & 0xFFFFu) << 16)); // td0[31:16]|td1[15:0]
  g1[3] = (int)((tile_rows >> 16) | ((tile_k & 0xFFFFu) << 16)); // td1[31:16]|tile_dim0
  g1[4] = (int)(tile_rows & 0xFFFFu);                            // tile_dim1, tile_dim2=0
  g1[5] = (int)(unsigned)row_stride_elems;                       // dim0_stride[31:0]
  g1[6] = (int)(unsigned)((row_stride_elems >> 32) & 0xFFFFull); // dim0_stride[47:32]
  g1[7] = 0;
  i32x4 z4 = {0, 0, 0, 0};
  i32x8 z8 = {0, 0, 0, 0, 0, 0, 0, 0};
  __builtin_amdgcn_tensor_load_to_lds(g0, g1, z4, z4, z8, 0);
}

__device__ __forceinline__ unsigned lds_off(const void* p) {
  return (unsigned)(unsigned long long)p;   // generic LDS ptr: addr[31:0] = LDS offset
}

// ---------------------------------------------------------------------------
// bf16 WMMA GEMM, TDM double-buffered:  C[M,N] = act(A[M,K] * Bt[N,K]^T + bias)
// Block = 128 thr = 4 waves -> 64x64 block tile; wave w owns rows w*16..w*16+15.
// Per K-step: wave0 DMAs A(64x32) + B(64x32) bf16 tiles into LDS via TDM while
// all waves run WMMA on the previous buffer.  Requires M%64==0, N%64==0, K%32==0.
// ACT: 0 none, 1 relu, 2 sigmoid.
// ---------------------------------------------------------------------------
template <int ACT, int BIAS, int OF32, int OB16, int OB16T>
__global__ __launch_bounds__(128)
void gemm_bf16_wmma(const __bf16* __restrict__ A, const __bf16* __restrict__ Bt,
                    const float* __restrict__ bias,
                    float* __restrict__ Cf, __bf16* __restrict__ Cb,
                    __bf16* __restrict__ CbT,
                    int M, int N, int K,
                    int lda, int ldb, int ldcf, int ldcb, int ldct) {
  __shared__ __attribute__((aligned(16))) __bf16 sA[2][64 * 32];
  __shared__ __attribute__((aligned(16))) __bf16 sB[2][64 * 32];
  int lane = threadIdx.x & 31;
  int wave = threadIdx.x >> 5;
  int nl = lane & 15, hf = lane >> 4;
  int mBlk = blockIdx.y * 64;
  int n0 = blockIdx.x * 64;

  const __bf16* Ab = A + (size_t)mBlk * lda;
  const __bf16* Bb = Bt + (size_t)n0 * ldb;
  int T = K / 32;

  if (wave == 0) {
    tdm_load_tile_bf16(lds_off(&sA[0][0]), Ab, 32, 64, (unsigned long long)lda);
    tdm_load_tile_bf16(lds_off(&sB[0][0]), Bb, 32, 64, (unsigned long long)ldb);
  }

  v8f acc[4] = {};
  for (int i = 0; i < T; ++i) {
    int buf = i & 1;
    __syncthreads();  // all waves done with buf^1 -> safe to refill
    if (wave == 0) {
      if (i + 1 < T) {
        tdm_load_tile_bf16(lds_off(&sA[buf ^ 1][0]), Ab + (size_t)(i + 1) * 32,
                           32, 64, (unsigned long long)lda);
        tdm_load_tile_bf16(lds_off(&sB[buf ^ 1][0]), Bb + (size_t)(i + 1) * 32,
                           32, 64, (unsigned long long)ldb);
        __builtin_amdgcn_s_wait_tensorcnt(2);  // TDM in-order: tile i landed
      } else {
        __builtin_amdgcn_s_wait_tensorcnt(0);
      }
    }
    __syncthreads();  // buf published to all waves

    // A fragment from LDS (2 x ds_load_b128)
    const __bf16* pa = &sA[buf][(wave * 16 + nl) * 32 + hf * 8];
    v8bf lo = *(const v8bf*)pa;
    v8bf hi = *(const v8bf*)(pa + 16);
    v16bf a = __builtin_shufflevector(lo, hi, 0,1,2,3,4,5,6,7,8,9,10,11,12,13,14,15);
#pragma unroll
    for (int j = 0; j < 4; ++j) {
      const __bf16* pb = &sB[buf][(j * 16 + nl) * 32 + hf * 8];
      v16bf b = __builtin_shufflevector(*(const v8bf*)pb, *(const v8bf*)(pb + 16), 0,1,2,3,4,5,6,7,8,9,10,11,12,13,14,15);
      acc[j] = wmma_bf16(a, b, acc[j]);
    }
  }

  int m0 = mBlk + wave * 16;
#pragma unroll
  for (int j = 0; j < 4; ++j) {
    int n = n0 + j * 16 + nl;
    float bv = BIAS ? bias[n] : 0.0f;
#pragma unroll
    for (int r = 0; r < 8; ++r) {
      int m = m0 + r + 8 * hf;  // D layout: m = vgpr + 8*half, n = lane&15
      float v = acc[j][r] + bv;
      if (ACT == 1) v = fmaxf(v, 0.0f);
      else if (ACT == 2) v = 1.0f / (1.0f + __expf(-v));
      if (OF32)  Cf[(size_t)m * ldcf + n] = v;
      if (OB16)  Cb[(size_t)m * ldcb + n] = (__bf16)v;
      if (OB16T) CbT[(size_t)n * ldct + m] = (__bf16)v;
    }
  }
}

// ---------------------------------------------------------------------------
// Flash attention: one wave per (b, h, 16-row q tile).  Streams 64-key tiles,
// online softmax, P bounced through 2KB LDS (D-layout -> A-layout reshape).
// ---------------------------------------------------------------------------
__global__ __launch_bounds__(32)
void flash_attn_kernel(const __bf16* __restrict__ q, const __bf16* __restrict__ k,
                       const __bf16* __restrict__ vT, float* __restrict__ out,
                       int N, int D, int ldvT) {
  __shared__ __attribute__((aligned(16))) __bf16 sP[16 * 64];
  int lane = threadIdx.x;
  int nl = lane & 15, hf = lane >> 4;
  int m0 = blockIdx.x * 16;
  int h  = blockIdx.y;
  int b  = blockIdx.z;

  const __bf16* qb = q + (size_t)b * N * D;
  const __bf16* kb = k + (size_t)b * N * D;

  v16bf aq0 = load_frag_a(qb, D, m0, h * 64, lane);
  v16bf aq1 = load_frag_a(qb, D, m0, h * 64 + 32, lane);

  float run_m[8], run_l[8];
  v8f acc[4] = {};
#pragma unroll
  for (int r = 0; r < 8; ++r) { run_m[r] = -3.0e38f; run_l[r] = 0.0f; }

  for (int kt = 0; kt < N / 64; ++kt) {
    int nk0 = kt * 64;
    int nk_pf = (nk0 + 64 < N) ? nk0 + 64 : nk0;
    __builtin_prefetch(kb + (size_t)(nk_pf + nl * 4) * D + h * 64, 0, 1);
    v8f s[4] = {};
#pragma unroll
    for (int j = 0; j < 4; ++j) {
      s[j] = wmma_bf16(aq0, load_frag_b(kb, D, nk0 + j * 16, h * 64, lane), s[j]);
      s[j] = wmma_bf16(aq1, load_frag_b(kb, D, nk0 + j * 16, h * 64 + 32, lane), s[j]);
      s[j] = s[j] * 0.125f;  // Dh^-0.5
    }
#pragma unroll
    for (int r = 0; r < 8; ++r) {
      float mx = fmaxf(fmaxf(s[0][r], s[1][r]), fmaxf(s[2][r], s[3][r]));
#pragma unroll
      for (int off = 1; off < 16; off <<= 1) mx = fmaxf(mx, __shfl_xor(mx, off, 32));
      float nm  = fmaxf(run_m[r], mx);
      float scl = __expf(run_m[r] - nm);
      run_m[r] = nm;
      float sm = 0.0f;
#pragma unroll
      for (int j = 0; j < 4; ++j) {
        float p = __expf(s[j][r] - nm);
        sm += p;
        sP[(r + 8 * hf) * 64 + j * 16 + nl] = (__bf16)p;
      }
#pragma unroll
      for (int off = 1; off < 16; off <<= 1) sm += __shfl_xor(sm, off, 32);
      run_l[r] = run_l[r] * scl + sm;
#pragma unroll
      for (int j = 0; j < 4; ++j) acc[j][r] *= scl;
    }
    __syncthreads();
    v16bf pa0, pa1;
    {
      const __bf16* p0 = &sP[(lane & 15) * 64 + hf * 8];
      v8bf lo = *(const v8bf*)p0;
      v8bf hi = *(const v8bf*)(p0 + 16);
      pa0 = __builtin_shufflevector(lo, hi, 0,1,2,3,4,5,6,7,8,9,10,11,12,13,14,15);
      const __bf16* p1 = &sP[(lane & 15) * 64 + 32 + hf * 8];
      v8bf lo1 = *(const v8bf*)p1;
      v8bf hi1 = *(const v8bf*)(p1 + 16);
      pa1 = __builtin_shufflevector(lo1, hi1, 0,1,2,3,4,5,6,7,8,9,10,11,12,13,14,15);
    }
    __syncthreads();
    const __bf16* vb = vT + (size_t)h * 64 * ldvT + (size_t)b * N + nk0;
#pragma unroll
    for (int j = 0; j < 4; ++j) {
      acc[j] = wmma_bf16(pa0, load_frag_b(vb, ldvT, j * 16, 0, lane), acc[j]);
      acc[j] = wmma_bf16(pa1, load_frag_b(vb, ldvT, j * 16, 32, lane), acc[j]);
    }
  }
#pragma unroll
  for (int j = 0; j < 4; ++j)
#pragma unroll
    for (int r = 0; r < 8; ++r) {
      int m = m0 + r + 8 * hf;
      out[((size_t)b * N + m) * D + h * 64 + j * 16 + nl] = acc[j][r] / run_l[r];
    }
}

// ---------------------------------------------------------------------------
// Elementwise / reduction helpers
// ---------------------------------------------------------------------------
__global__ void to_bf16_kernel(const float* __restrict__ s, __bf16* __restrict__ d, int n) {
  int i = blockIdx.x * 256 + threadIdx.x;
  if (i < n) d[i] = (__bf16)s[i];
}

__global__ void transpose_to_bf16(const float* __restrict__ W, __bf16* __restrict__ Wt,
                                  int K, int N) {
  __shared__ float tile[16][17];
  int kb = blockIdx.y * 16, nb = blockIdx.x * 16;
  int tx = threadIdx.x, ty = threadIdx.y;
  tile[ty][tx] = W[(size_t)(kb + ty) * N + nb + tx];
  __syncthreads();
  Wt[(size_t)(nb + ty) * K + kb + tx] = (__bf16)tile[tx][ty];
}

__global__ void rowsumsq_kernel(const float* __restrict__ x, float* __restrict__ out, int cols) {
  int row = blockIdx.x;
  const float* p = x + (size_t)row * cols;
  float s = 0.0f;
  for (int c = threadIdx.x; c < cols; c += 32) { float v = p[c]; s += v * v; }
  for (int off = 16; off; off >>= 1) s += __shfl_xor(s, off, 32);
  if (threadIdx.x == 0) out[row] = s;
}

__global__ void rowsum_bf16_kernel(const __bf16* __restrict__ a, float* __restrict__ out, int cols) {
  int row = blockIdx.x;
  const __bf16* p = a + (size_t)row * cols;
  float s = 0.0f;
  for (int c = threadIdx.x; c < cols; c += 32) s += (float)p[c];
  for (int off = 16; off; off >>= 1) s += __shfl_xor(s, off, 32);
  if (threadIdx.x == 0) out[row] = fmaxf(s, 1e-8f);
}

__global__ void infl_kernel(const float* __restrict__ xs, const float* __restrict__ x2,
                            const float* __restrict__ s2, const float* __restrict__ scales,
                            const float* __restrict__ imp,
                            __bf16* __restrict__ infl, __bf16* __restrict__ inflT,
                            int Ntok, int S) {
  int row = blockIdx.x;  // b*Ntok + n
  int s = threadIdx.x;   // blockDim.x == S
  int b = row / Ntok, n = row % Ntok;
  float d2 = fmaxf(x2[row] + s2[s] - 2.0f * xs[(size_t)row * S + s], 0.0f);
  float sc = fmaxf(fabsf(scales[s]), 1e-6f);
  float v = __expf(-0.5f * d2 / (sc * sc)) * fabsf(imp[s]);
  infl[(size_t)row * S + s] = (__bf16)v;
  inflT[((size_t)b * S + s) * Ntok + n] = (__bf16)v;
}

__global__ void norm_to_bf16_kernel(const float* __restrict__ g, const float* __restrict__ total,
                                    __bf16* __restrict__ out, int cols, int n) {
  int i = blockIdx.x * 256 + threadIdx.x;
  if (i < n) out[i] = (__bf16)(g[i] / total[i / cols]);
}

__global__ void splat_trans_kernel(const float* __restrict__ enc, const float* __restrict__ Wt,
                                   const float* __restrict__ bt, __bf16* __restrict__ trans,
                                   int S, int Db) {
  int idx = blockIdx.x * 256 + threadIdx.x;
  int e = idx % Db;
  int bs = idx / Db;
  int s = bs % S;
  const float* er = enc + (size_t)bs * Db;
  const float* w  = Wt + (size_t)s * Db * Db + e;
  float acc = bt[(size_t)s * Db + e];
  for (int d = 0; d < Db; ++d) acc += er[d] * w[(size_t)d * Db];
  trans[(size_t)bs * Db + e] = (__bf16)acc;
}

__global__ void blend_to_bf16_kernel(const float* __restrict__ so, const float* __restrict__ fl,
                                     const float* __restrict__ gt, __bf16* __restrict__ out, int n) {
  int i = blockIdx.x * 256 + threadIdx.x;
  if (i < n) out[i] = (__bf16)(0.6f * so[i] + 0.4f * fl[i] * gt[i]);
}

// ---------------------------------------------------------------------------
// Launch
// ---------------------------------------------------------------------------
extern "C" void kernel_launch(void* const* d_in, const int* in_sizes, int n_in,
                              void* d_out, int out_size, void* d_ws, size_t ws_size,
                              hipStream_t stream) {
  const int B = 2, N = 1024, D = 1024, S = 64, Db = 512, H = 16;
  const float* x    = (const float*)d_in[0];
  const float* Wq   = (const float*)d_in[1];
  const float* Wk   = (const float*)d_in[2];
  const float* Wv   = (const float*)d_in[3];
  const float* Wo   = (const float*)d_in[4];
  const float* spos = (const float*)d_in[5];
  const float* sscl = (const float*)d_in[6];
  const float* simp = (const float*)d_in[7];
  const float* We   = (const float*)d_in[8];
  const float* be   = (const float*)d_in[9];
  const float* Wt   = (const float*)d_in[10];
  const float* bt   = (const float*)d_in[11];
  const float* Wd   = (const float*)d_in[12];
  const float* bd   = (const float*)d_in[13];
  const float* Wg   = (const float*)d_in[14];
  const float* bg   = (const float*)d_in[15];
  float* out = (float*)d_out;

  char* wp = (char*)d_ws;
  auto carve = [&](size_t bytes) -> void* {
    void* r = (void*)wp;
    wp += (bytes + 255) & ~(size_t)255;
    return r;
  };
  __bf16* xb    = (__bf16*)carve((size_t)B * N * D * 2);
  __bf16* xT    = (__bf16*)carve((size_t)B * D * N * 2);
  __bf16* WqT   = (__bf16*)carve((size_t)D * D * 2);
  __bf16* WkT   = (__bf16*)carve((size_t)D * D * 2);
  __bf16* WvT   = (__bf16*)carve((size_t)D * D * 2);
  __bf16* WoT   = (__bf16*)carve((size_t)D * D * 2);
  __bf16* WgT   = (__bf16*)carve((size_t)D * D * 2);
  __bf16* WeT   = (__bf16*)carve((size_t)Db * D * 2);
  __bf16* WdT   = (__bf16*)carve((size_t)D * Db * 2);
  __bf16* spb   = (__bf16*)carve((size_t)S * D * 2);
  __bf16* qbuf  = (__bf16*)carve((size_t)B * N * D * 2);
  __bf16* kbuf  = (__bf16*)carve((size_t)B * N * D * 2);
  __bf16* vT    = (__bf16*)carve((size_t)D * B * N * 2);   // [D, B*N]
  float*  gate  = (float*) carve((size_t)B * N * D * 4);
  float*  xs    = (float*) carve((size_t)B * N * S * 4);
  float*  x2    = (float*) carve((size_t)B * N * 4);
  float*  s2v   = (float*) carve((size_t)S * 4);
  __bf16* inflb = (__bf16*)carve((size_t)B * N * S * 2);
  __bf16* inflT = (__bf16*)carve((size_t)B * S * N * 2);
  float*  totl  = (float*) carve((size_t)B * S * 4);
  float*  gath  = (float*) carve((size_t)B * S * D * 4);
  __bf16* gathb = (__bf16*)carve((size_t)B * S * D * 2);
  float*  encv  = (float*) carve((size_t)B * S * Db * 4);
  __bf16* trnb  = (__bf16*)carve((size_t)B * S * Db * 2);
  __bf16* decT  = (__bf16*)carve((size_t)D * B * S * 2);   // [D, B*S]
  float*  flow  = (float*) carve((size_t)B * N * D * 4);
  float*  sout  = (float*) carve((size_t)B * N * D * 4);
  __bf16* blnd  = (__bf16*)carve((size_t)B * N * D * 2);

  dim3 t16(16, 16);
  to_bf16_kernel<<<(B * N * D + 255) / 256, 256, 0, stream>>>(x, xb, B * N * D);
  to_bf16_kernel<<<(S * D + 255) / 256, 256, 0, stream>>>(spos, spb, S * D);
  transpose_to_bf16<<<dim3(D / 16, D / 16), t16, 0, stream>>>(Wq, WqT, D, D);
  transpose_to_bf16<<<dim3(D / 16, D / 16), t16, 0, stream>>>(Wk, WkT, D, D);
  transpose_to_bf16<<<dim3(D / 16, D / 16), t16, 0, stream>>>(Wv, WvT, D, D);
  transpose_to_bf16<<<dim3(D / 16, D / 16), t16, 0, stream>>>(Wo, WoT, D, D);
  transpose_to_bf16<<<dim3(D / 16, D / 16), t16, 0, stream>>>(Wg, WgT, D, D);
  transpose_to_bf16<<<dim3(Db / 16, D / 16), t16, 0, stream>>>(We, WeT, D, Db);
  transpose_to_bf16<<<dim3(D / 16, Db / 16), t16, 0, stream>>>(Wd, WdT, Db, D);
  for (int b = 0; b < B; ++b)
    transpose_to_bf16<<<dim3(D / 16, N / 16), t16, 0, stream>>>(
        x + (size_t)b * N * D, xT + (size_t)b * D * N, N, D);

#define GEMM(ACT, BIAS, OF, OB, OBT, Ap, Bp, bp, Cfp, Cbp, CbTp, M_, N_, K_, lda_, ldb_, ldcf_, ldcb_, ldct_) \
  gemm_bf16_wmma<ACT, BIAS, OF, OB, OBT><<<dim3((N_) / 64, (M_) / 64), 128, 0, stream>>>( \
      Ap, Bp, bp, Cfp, Cbp, CbTp, M_, N_, K_, lda_, ldb_, ldcf_, ldcb_, ldct_)

  // ---- projections ----
  GEMM(0, 0, 0, 1, 0, xb, WqT, nullptr, nullptr, qbuf, nullptr, B * N, D, D, D, D, 0, D, 0);
  GEMM(0, 0, 0, 1, 0, xb, WkT, nullptr, nullptr, kbuf, nullptr, B * N, D, D, D, D, 0, D, 0);
  GEMM(0, 0, 0, 0, 1, xb, WvT, nullptr, nullptr, nullptr, vT, B * N, D, D, D, D, 0, 0, B * N);
  GEMM(2, 1, 1, 0, 0, xb, WgT, bg, gate, nullptr, nullptr, B * N, D, D, D, D, D, 0, 0);

  // ---- attention (flash, WMMA) ----
  flash_attn_kernel<<<dim3(N / 16, H, B), 32, 0, stream>>>(qbuf, kbuf, vT, sout, N, D, B * N);

  // ---- splat RBF influences ----
  GEMM(0, 0, 1, 0, 0, xb, spb, nullptr, xs, nullptr, nullptr, B * N, S, D, D, D, S, 0, 0);
  rowsumsq_kernel<<<B * N, 32, 0, stream>>>(x, x2, D);
  rowsumsq_kernel<<<S, 32, 0, stream>>>(spos, s2v, D);
  infl_kernel<<<B * N, S, 0, stream>>>(xs, x2, s2v, sscl, simp, inflb, inflT, N, S);
  rowsum_bf16_kernel<<<B * S, 32, 0, stream>>>(inflT, totl, N);

  // ---- gathered = inflT @ x (per batch), normalized ----
  for (int b = 0; b < B; ++b)
    GEMM(0, 0, 1, 0, 0, inflT + (size_t)b * S * N, xT + (size_t)b * D * N, nullptr,
         gath + (size_t)b * S * D, nullptr, nullptr, S, D, N, N, N, D, 0, 0);
  norm_to_bf16_kernel<<<(B * S * D + 255) / 256, 256, 0, stream>>>(gath, totl, gathb, D, B * S * D);

  // ---- enc = relu(gathered @ We + be) ----
  GEMM(1, 1, 1, 0, 0, gathb, WeT, be, encv, nullptr, nullptr, B * S, Db, D, D, D, Db, 0, 0);
  // ---- per-splat transform (tiny, VALU) ----
  splat_trans_kernel<<<(B * S * Db + 255) / 256, 256, 0, stream>>>(encv, Wt, bt, trnb, S, Db);
  // ---- dec = trans @ Wd + bd, stored transposed for the flow GEMM ----
  GEMM(0, 1, 0, 0, 1, trnb, WdT, bd, nullptr, nullptr, decT, B * S, D, Db, Db, Db, 0, 0, B * S);
  // ---- flow = infl @ dec (per batch) ----
  for (int b = 0; b < B; ++b)
    GEMM(0, 0, 1, 0, 0, inflb + (size_t)b * N * S, decT + (size_t)b * S, nullptr,
         flow + (size_t)b * N * D, nullptr, nullptr, N, D, S, S, B * S, D, 0, 0);

  // ---- blend and final projection ----
  blend_to_bf16_kernel<<<(B * N * D + 255) / 256, 256, 0, stream>>>(sout, flow, gate, blnd, B * N * D);
  GEMM(0, 0, 1, 0, 0, blnd, WoT, nullptr, out, nullptr, nullptr, B * N, D, D, D, D, D, 0, 0);
#undef GEMM
}
